// LorentzSpatialBlock_2293512536884
// MI455X (gfx1250) — matrix operand order repaired and checked by reading, not compiled, not run on hardware
//
#include <hip/hip_runtime.h>
#include <hip/hip_bf16.h>
#include <math.h>

typedef _Float16 f16;
typedef __attribute__((ext_vector_type(16))) _Float16 v16h;
typedef __attribute__((ext_vector_type(8)))  _Float16 v8h;
typedef __attribute__((ext_vector_type(8)))  float    v8f;

#define ROWS 48      // tokens per workgroup (2 batch rows * J=24)
#define DMODEL 256
#define J_TOK 24

// ---------------------------------------------------------------------------
// Weight packing: f32 row-major (K x N)  ->  f16 in WMMA B-operand lane order.
// Tile (kt, nt) of 32x16; within a tile: element index = lane*16 + h where
//   row = kt*32 + (h/8)*16 + (lane/16)*8 + h%8,  col = nt*16 + (lane%16)
// Tiles stored k-major within each n-tile: tile_id = nt*(K/32) + kt.
// ---------------------------------------------------------------------------
__global__ void pack_w_kernel(const float* __restrict__ src, f16* __restrict__ dst,
                              int K, int N) {
  int idx = blockIdx.x * 256 + threadIdx.x;
  int total = K * N;
  if (idx >= total) return;
  int KT = K >> 5;
  int perNT = KT * 512;
  int nt = idx / perNT;
  int r1 = idx - nt * perNT;
  int kt = r1 >> 9;
  int r2 = r1 & 511;
  int lane = r2 >> 4;
  int h = r2 & 15;
  int row = kt * 32 + ((h >> 3) << 4) + ((lane >> 4) << 3) + (h & 7);
  int col = nt * 16 + (lane & 15);
  dst[idx] = (f16)src[row * N + col];
}

// ---------------------------------------------------------------------------
// WMMA helpers
// ---------------------------------------------------------------------------
__device__ __forceinline__ float wred_sum(float v) {
#pragma unroll
  for (int m = 16; m >= 1; m >>= 1) v += __shfl_xor(v, m, 32);
  return v;
}

// A operand (16x32 f16 tile) from row-major LDS buffer, per ISA 16-bit A layout:
// lane L holds row M = L%16; halfs h -> K = kt*32 + (h/8)*16 + (L/16)*8 + h%8
__device__ __forceinline__ v16h load_A(const f16* __restrict__ A, int ld,
                                       int mt, int kt, int lane) {
  const int row = mt * 16 + (lane & 15);
  const int k0  = kt * 32 + ((lane >> 4) << 3);
  union { v16h v; v8h h[2]; } u;
  u.h[0] = *(const v8h*)(A + row * ld + k0);
  u.h[1] = *(const v8h*)(A + row * ld + k0 + 16);
  return u.v;
}

// Two N-tiles per A fragment: halves A-operand LDS traffic, two independent
// accumulation chains pipeline through the WMMA unit.
template <int KT>
__device__ __forceinline__ void gemm_tile2(const f16* __restrict__ A, int ldA,
                                           const v16h* __restrict__ Bpk,
                                           int mt, int ntp, int lane,
                                           v8f& acc0, v8f& acc1) {
#pragma unroll
  for (int kt = 0; kt < KT; ++kt) {
    v16h a  = load_A(A, ldA, mt, kt, lane);
    v16h b0 = Bpk[((2 * ntp + 0) * KT + kt) * 32 + lane];
    v16h b1 = Bpk[((2 * ntp + 1) * KT + kt) * 32 + lane];
    acc0 = __builtin_amdgcn_wmma_f32_16x16x32_f16(false, a, false, b0,
                                                  (short)0, acc0, false, false);
    acc1 = __builtin_amdgcn_wmma_f32_16x16x32_f16(false, a, false, b1,
                                                  (short)0, acc1, false, false);
  }
}

// C/D layout: vgpr r, lane L -> M = mt*16 + (L/16)*8 + r, N = nt*16 + L%16
__device__ __forceinline__ void store_tile16(f16* dst, int ld, int mt, int nt,
                                             int lane, v8f acc) {
  const int mb = mt * 16 + ((lane >> 4) << 3);
  const int n  = nt * 16 + (lane & 15);
#pragma unroll
  for (int r = 0; r < 8; ++r) dst[(mb + r) * ld + n] = (f16)acc[r];
}

__device__ __forceinline__ void store_tile32(float* dst, int ld, int mt, int nt,
                                             int lane, v8f acc) {
  const int mb = mt * 16 + ((lane >> 4) << 3);
  const int n  = nt * 16 + (lane & 15);
#pragma unroll
  for (int r = 0; r < 8; ++r) dst[(mb + r) * ld + n] = acc[r];
}

// ---------------------------------------------------------------------------
// Fully fused Lorentz transformer block. One WG handles 2 batch rows (48 tok).
// LDS footprint 144KB -> 2 workgroups per 320KB WGP (4 waves/SIMD).
// ---------------------------------------------------------------------------
__global__ __launch_bounds__(256)
void lorentz_block_kernel(const float* __restrict__ x,      // (N, 257)
                          const float* __restrict__ vtan,   // (N, 256)
                          const float* __restrict__ topo,   // (24, 24)
                          const float* __restrict__ n1g, const float* __restrict__ n1b,
                          const float* __restrict__ nvg, const float* __restrict__ nvb,
                          const float* __restrict__ n2g, const float* __restrict__ n2b,
                          const float* __restrict__ b1,  const float* __restrict__ b2,
                          const v16h* __restrict__ Wq,  const v16h* __restrict__ Wk,
                          const v16h* __restrict__ Wv,  const v16h* __restrict__ Wvk,
                          const v16h* __restrict__ Wo,  const v16h* __restrict__ W1,
                          const v16h* __restrict__ W2,
                          float* __restrict__ out) {       // (N, 257)
  const int tid  = threadIdx.x;
  const int lane = tid & 31;
  const int wave = tid >> 5;
  const long row0 = (long)blockIdx.x * ROWS;

  extern __shared__ __align__(16) char smem[];
  f16*   s_ut  = (f16*)(smem);               //  0K..24K  LN1 tangent (later: attn out o)
  f16*   s_vn  = (f16*)(smem + 24 * 1024);   // 24K..48K  LN(v_tan)   (later: t for FFN)
  f16*   s_q   = (f16*)(smem + 48 * 1024);   // 48K..72K
  f16*   s_k   = (f16*)(smem + 72 * 1024);   // 72K..96K
  f16*   s_v   = (f16*)(smem + 96 * 1024);   // 96K..120K
  f16*   s_o   = s_ut;                       // attn output reuses ut slot
  f16*   s_t   = s_vn;                       // FFN input reuses vn slot
  float* s_o2  = (float*)(smem + 48 * 1024); // 48KB f32, reuses q/k slots
  f16*   s_hid = (f16*)(smem + 48 * 1024);   // 48K..144K FFN hidden (48x1024 f16)
  float* s_t2  = (float*)(smem);             // 48KB f32, reuses ut/vn slots

  // ---- Phase 1: ut = LN1(logmap0(x)); vn = LNv(v_tan) --------------------
  for (int r = wave; r < ROWS; r += 8) {
    const long grow = row0 + r;
    const float* xr = x + grow * 257;
    const float* vr = vtan + grow * 256;
    const int e0 = lane * 8;
    float xs[8], vt[8];
    float s1 = 0.f, s2 = 0.f, p1 = 0.f, p2 = 0.f;
#pragma unroll
    for (int i = 0; i < 8; ++i) {
      xs[i] = xr[1 + e0 + i];
      s1 += xs[i]; s2 += xs[i] * xs[i];
    }
    const float4 va = *(const float4*)(vr + e0);
    const float4 vb = *(const float4*)(vr + e0 + 4);
    vt[0]=va.x; vt[1]=va.y; vt[2]=va.z; vt[3]=va.w;
    vt[4]=vb.x; vt[5]=vb.y; vt[6]=vb.z; vt[7]=vb.w;
#pragma unroll
    for (int i = 0; i < 8; ++i) { p1 += vt[i]; p2 += vt[i] * vt[i]; }
    s1 = wred_sum(s1); s2 = wred_sum(s2);
    p1 = wred_sum(p1); p2 = wred_sum(p2);

    const float nn  = sqrtf(fmaxf(s2, 1e-12f));
    const float x0  = fmaxf(xr[0], 1.0f + 1e-7f);
    const float ach = logf(x0 + sqrtf(x0 * x0 - 1.0f));   // arccosh(x0)
    const float sc  = ach / nn;                           // u = sc * xs
    const float muU = sc * s1 * (1.0f / 256.0f);
    const float m2U = sc * sc * s2 * (1.0f / 256.0f);
    const float rsU = rsqrtf(fmaxf(m2U - muU * muU, 0.0f) + 1e-5f);
    const float muV = p1 * (1.0f / 256.0f);
    const float rsV = rsqrtf(fmaxf(p2 * (1.0f / 256.0f) - muV * muV, 0.0f) + 1e-5f);
#pragma unroll
    for (int i = 0; i < 8; ++i) {
      const float u = sc * xs[i];
      s_ut[r * DMODEL + e0 + i] = (f16)((u - muU) * rsU * n1g[e0 + i] + n1b[e0 + i]);
      s_vn[r * DMODEL + e0 + i] = (f16)((vt[i] - muV) * rsV * nvg[e0 + i] + nvb[e0 + i]);
    }
  }
  __syncthreads();

  // ---- Phase 2: q = ut@Wq, k = ut@Wk + vn@Wvk, v = ut@Wv (WMMA) ----------
  // 72 jobs: outm(3) x mt(3) x ntpair(8); 9 per wave, wave-uniform.
  for (int job = wave; job < 72; job += 8) {
    const int outm = job / 24;
    const int rem  = job % 24;
    const int mt = rem >> 3, ntp = rem & 7;
    v8f a0 = {}, a1 = {};
    const v16h* Bp = (outm == 0) ? Wq : (outm == 1) ? Wk : Wv;
    gemm_tile2<8>(s_ut, DMODEL, Bp, mt, ntp, lane, a0, a1);
    if (outm == 1) gemm_tile2<8>(s_vn, DMODEL, Wvk, mt, ntp, lane, a0, a1);
    f16* dst = (outm == 0) ? s_q : (outm == 1) ? s_k : s_v;
    store_tile16(dst, DMODEL, mt, 2 * ntp + 0, lane, a0);
    store_tile16(dst, DMODEL, mt, 2 * ntp + 1, lane, a1);
  }
  __syncthreads();

  // ---- Phase 3: attention (J=24, dh=32), one thread per (b,h,i) row ------
  for (int t = tid; t < 2 * 8 * J_TOK; t += 256) {
    const int bb = t / 192;
    const int rm = t % 192;
    const int h  = rm / J_TOK;
    const int i  = rm % J_TOK;
    const int qoff = (bb * J_TOK + i) * DMODEL + h * 32;
    v8h qv[4];
#pragma unroll
    for (int c = 0; c < 4; ++c) qv[c] = *(const v8h*)(s_q + qoff + c * 8);
    float scores[J_TOK];
    float mx = -3.0e38f;
#pragma unroll
    for (int j = 0; j < J_TOK; ++j) {
      const int koff = (bb * J_TOK + j) * DMODEL + h * 32;
      float d = 0.f;
#pragma unroll
      for (int c = 0; c < 4; ++c) {
        v8h kv = *(const v8h*)(s_k + koff + c * 8);
#pragma unroll
        for (int e = 0; e < 8; ++e) d += (float)qv[c][e] * (float)kv[e];
      }
      d = d * 0.17677669529663687f + topo[i * J_TOK + j];  // 1/sqrt(32)
      scores[j] = d;
      mx = fmaxf(mx, d);
    }
    float ssum = 0.f;
#pragma unroll
    for (int j = 0; j < J_TOK; ++j) {
      const float p = __expf(scores[j] - mx);
      scores[j] = p; ssum += p;
    }
    const float inv = 1.0f / ssum;
    float oacc[32];
#pragma unroll
    for (int e = 0; e < 32; ++e) oacc[e] = 0.f;
#pragma unroll
    for (int j = 0; j < J_TOK; ++j) {
      const float p = scores[j] * inv;
      const int voff = (bb * J_TOK + j) * DMODEL + h * 32;
#pragma unroll
      for (int c = 0; c < 4; ++c) {
        v8h vv = *(const v8h*)(s_v + voff + c * 8);
#pragma unroll
        for (int e = 0; e < 8; ++e) oacc[c * 8 + e] += p * (float)vv[e];
      }
    }
#pragma unroll
    for (int e = 0; e < 32; ++e) s_o[qoff + e] = (f16)oacc[e];
  }
  __syncthreads();

  // ---- Phase 4: o2 = o @ Wo (WMMA, f32 result to LDS) --------------------
  for (int job = wave; job < 24; job += 8) {
    const int mt = job >> 3, ntp = job & 7;
    v8f a0 = {}, a1 = {};
    gemm_tile2<8>(s_o, DMODEL, Wo, mt, ntp, lane, a0, a1);
    store_tile32(s_o2, DMODEL, mt, 2 * ntp + 0, lane, a0);
    store_tile32(s_o2, DMODEL, mt, 2 * ntp + 1, lane, a1);
  }
  __syncthreads();

  // ---- Phase 4b: t = LN2(logmap0(x) + o2)  (recompute u from x, f32) -----
  for (int r = wave; r < ROWS; r += 8) {
    const long grow = row0 + r;
    const float* xr = x + grow * 257;
    const int e0 = lane * 8;
    float xs[8]; float s2 = 0.f;
#pragma unroll
    for (int i = 0; i < 8; ++i) { xs[i] = xr[1 + e0 + i]; s2 += xs[i] * xs[i]; }
    s2 = wred_sum(s2);
    const float nn  = sqrtf(fmaxf(s2, 1e-12f));
    const float x0  = fmaxf(xr[0], 1.0f + 1e-7f);
    const float ach = logf(x0 + sqrtf(x0 * x0 - 1.0f));
    const float sc  = ach / nn;
    float rv[8]; float t1 = 0.f, t2 = 0.f;
#pragma unroll
    for (int i = 0; i < 8; ++i) {
      const float val = sc * xs[i] + s_o2[r * DMODEL + e0 + i];
      rv[i] = val; t1 += val; t2 += val * val;
    }
    t1 = wred_sum(t1); t2 = wred_sum(t2);
    const float mu = t1 * (1.0f / 256.0f);
    const float rs = rsqrtf(fmaxf(t2 * (1.0f / 256.0f) - mu * mu, 0.0f) + 1e-5f);
#pragma unroll
    for (int i = 0; i < 8; ++i)
      s_t[r * DMODEL + e0 + i] = (f16)((rv[i] - mu) * rs * n2g[e0 + i] + n2b[e0 + i]);
  }
  __syncthreads();

  // ---- Phase 5: hidden = gelu(t @ W1 + b1)  (WMMA, N=1024) ---------------
  for (int job = wave; job < 96; job += 8) {
    const int mt  = job / 32;
    const int ntp = job % 32;
    v8f a0 = {}, a1 = {};
    gemm_tile2<8>(s_t, DMODEL, W1, mt, ntp, lane, a0, a1);
    const int mb = mt * 16 + ((lane >> 4) << 3);
#pragma unroll
    for (int half = 0; half < 2; ++half) {
      const int nt = 2 * ntp + half;
      const int n  = nt * 16 + (lane & 15);
      const float bb1 = b1[n];
      v8f acc = half ? a1 : a0;
#pragma unroll
      for (int r8 = 0; r8 < 8; ++r8) {
        const float xg = acc[r8] + bb1;
        const float g  = 0.5f * xg *
            (1.0f + tanhf(0.7978845608028654f * (xg + 0.044715f * xg * xg * xg)));
        s_hid[(mb + r8) * 1024 + n] = (f16)g;
      }
    }
  }
  __syncthreads();

  // ---- Phase 6: t2 = hidden @ W2 + b2  (WMMA, K=1024) --------------------
  for (int job = wave; job < 24; job += 8) {
    const int mt = job >> 3, ntp = job & 7;
    v8f a0 = {}, a1 = {};
    gemm_tile2<32>(s_hid, 1024, W2, mt, ntp, lane, a0, a1);
    const int mb = mt * 16 + ((lane >> 4) << 3);
#pragma unroll
    for (int half = 0; half < 2; ++half) {
      const int n = (2 * ntp + half) * 16 + (lane & 15);
      const float bb2 = b2[n];
      v8f acc = half ? a1 : a0;
#pragma unroll
      for (int r8 = 0; r8 < 8; ++r8) s_t2[(mb + r8) * DMODEL + n] = acc[r8] + bb2;
    }
  }
  __syncthreads();

  // ---- Phase 7: out = expmap0(t2)  (non-temporal stream-out) -------------
  for (int r = wave; r < ROWS; r += 8) {
    const long grow = row0 + r;
    const int e0 = lane * 8;
    float tv[8]; float s2 = 0.f;
#pragma unroll
    for (int i = 0; i < 8; ++i) {
      tv[i] = s_t2[r * DMODEL + e0 + i];
      s2 += tv[i] * tv[i];
    }
    s2 = wred_sum(s2);
    const float nn  = sqrtf(fmaxf(s2, 1e-12f));
    const float en  = __expf(nn);
    const float ei  = 1.0f / en;
    const float ch  = 0.5f * (en + ei);
    const float shn = 0.5f * (en - ei) / nn;   // sinh(n)/n
    float* orow = out + grow * 257;
    if (lane == 0) __builtin_nontemporal_store(ch, orow);
#pragma unroll
    for (int i = 0; i < 8; ++i)
      __builtin_nontemporal_store(shn * tv[i], orow + 1 + e0 + i);
  }
}

// ---------------------------------------------------------------------------
extern "C" void kernel_launch(void* const* d_in, const int* in_sizes, int n_in,
                              void* d_out, int out_size, void* d_ws, size_t ws_size,
                              hipStream_t stream) {
  (void)in_sizes; (void)n_in; (void)out_size; (void)ws_size;
  const float* x    = (const float*)d_in[0];
  const float* vtan = (const float*)d_in[1];
  const float* topo = (const float*)d_in[2];
  const float* n1g  = (const float*)d_in[3];
  const float* n1b  = (const float*)d_in[4];
  const float* nvg  = (const float*)d_in[5];
  const float* nvb  = (const float*)d_in[6];
  const float* n2g  = (const float*)d_in[7];
  const float* n2b  = (const float*)d_in[8];
  const float* Wq   = (const float*)d_in[9];
  const float* Wk   = (const float*)d_in[10];
  const float* Wv   = (const float*)d_in[11];
  const float* Wvk  = (const float*)d_in[12];
  const float* Wo   = (const float*)d_in[13];
  const float* W1   = (const float*)d_in[14];
  const float* b1   = (const float*)d_in[15];
  const float* W2   = (const float*)d_in[16];
  const float* b2   = (const float*)d_in[17];

  // Packed f16 weights in workspace (total 1,703,936 bytes)
  f16* wsp  = (f16*)d_ws;
  f16* pWq  = wsp;
  f16* pWk  = pWq  + 65536;
  f16* pWv  = pWk  + 65536;
  f16* pWvk = pWv  + 65536;
  f16* pWo  = pWvk + 65536;
  f16* pW1  = pWo  + 65536;   // 256*1024
  f16* pW2  = pW1  + 262144;  // 1024*256

  pack_w_kernel<<<65536 / 256, 256, 0, stream>>>(Wq,  pWq,  256, 256);
  pack_w_kernel<<<65536 / 256, 256, 0, stream>>>(Wk,  pWk,  256, 256);
  pack_w_kernel<<<65536 / 256, 256, 0, stream>>>(Wv,  pWv,  256, 256);
  pack_w_kernel<<<65536 / 256, 256, 0, stream>>>(Wvk, pWvk, 256, 256);
  pack_w_kernel<<<65536 / 256, 256, 0, stream>>>(Wo,  pWo,  256, 256);
  pack_w_kernel<<<262144 / 256, 256, 0, stream>>>(W1, pW1,  256, 1024);
  pack_w_kernel<<<262144 / 256, 256, 0, stream>>>(W2, pW2, 1024, 256);

  const int smem_bytes = 144 * 1024;   // 2 workgroups fit per 320KB WGP
  hipFuncSetAttribute((const void*)lorentz_block_kernel,
                      hipFuncAttributeMaxDynamicSharedMemorySize, smem_bytes);

  // 98304 tokens / 48 per WG = 2048 workgroups of 256 threads (8 wave32s)
  lorentz_block_kernel<<<2048, 256, smem_bytes, stream>>>(
      x, vtan, topo, n1g, n1b, nvg, nvb, n2g, n2b, b1, b2,
      (const v16h*)pWq, (const v16h*)pWk, (const v16h*)pWv, (const v16h*)pWvk,
      (const v16h*)pWo, (const v16h*)pW1, (const v16h*)pW2, (float*)d_out);
}
